// DynamicQuantizedLinear_2104533975504
// MI455X (gfx1250) — matrix operand order repaired and checked by reading, not compiled, not run on hardware
//
#include <hip/hip_runtime.h>
#include <stdint.h>

typedef __attribute__((ext_vector_type(16))) _Float16 v16h;
typedef __attribute__((ext_vector_type(8)))  _Float16 v8h;
typedef __attribute__((ext_vector_type(2)))  _Float16 v2h;
typedef __attribute__((ext_vector_type(8)))  float    v8f;
typedef __attribute__((ext_vector_type(4)))  uint32_t v4u;

#define OUT_F 11008
#define IN_F  4096
#define GS    128
#define NG    32
#define MB    64

// fp32 x -> fp16 xh (one-time; 1 MB read, negligible vs 45 MB weight stream)
__global__ void xcvt_kernel(const float* __restrict__ x, _Float16* __restrict__ xh, int n) {
    int i = blockIdx.x * blockDim.x + threadIdx.x;
    if (i < n) xh[i] = (_Float16)x[i];
}

// Exact int8 -> f16 pair: (0x6400 | (w^0x80)) encodes w+1152 in f16 (unit-step range),
// packed add of -1152.0 recovers w exactly.
__device__ __forceinline__ v2h cvt2(uint32_t spread /* {00,b1,00,b0} */) {
    const v2h neg1152 = { (_Float16)(-1152.0f), (_Float16)(-1152.0f) };
    return __builtin_bit_cast(v2h, spread ^ 0x64806480u) + neg1152;
}

// Block = one N-tile (16 output features), 4 waves split the K reduction
// (8 groups each); deterministic LDS reduction; grid = 11008/16 = 688 blocks.
__global__ __launch_bounds__(128) void qlinear_wmma_kernel(
    const _Float16* __restrict__ xh,     // [64, 4096] fp16
    const uint8_t* __restrict__ wq,      // [11008, 4096] int8 (raw bytes)
    const _Float16* __restrict__ scale,  // [11008, 32] fp16
    const _Float16* __restrict__ bias,   // [11008] fp16
    const _Float16* __restrict__ resid,  // [11008] fp16
    float* __restrict__ out)             // [64, 11008] fp32
{
    const int lane  = threadIdx.x & 31;
    const int wave  = threadIdx.x >> 5;
    const int nbase = blockIdx.x * 16;
    const int col   = lane & 15;   // B/C layout: N = lane%16
    const int hi    = lane >> 4;   // half-wave K split
    const int o     = nbase + col; // this lane's output feature (weight row)

    const uint8_t*  wrow = wq + (size_t)o * IN_F;
    const _Float16* srow = scale + (size_t)o * NG;

    const int aoff = hi * 16;  // A: lanes 0-15 hold K k0..k0+15; lanes 16-31 k0+16..+31
    const int boff = hi * 16;  // B: lanes 0-15 hold K k0..k0+15; lanes 16-31 k0+16..+31

    v8f acc[4] = {};

    const int g0 = wave * (NG / 4);
    for (int g = g0; g < g0 + NG / 4; ++g) {
        __builtin_prefetch(wrow + (g + 1) * GS, 0, 0);  // speculative; dropped past end

        v8f gacc[4] = {};
        #pragma unroll
        for (int kk = 0; kk < 4; ++kk) {
            const int k0 = g * GS + kk * 32;

            // B fragment: 16 contiguous int8 -> 16 exact f16 (3 VALU per 2 elems)
            v4u wd = *(const v4u*)(wrow + k0 + boff);
            union { uint32_t u[8]; v16h v; } B;
            #pragma unroll
            for (int j = 0; j < 4; ++j) {
                uint32_t d  = wd[j];
                uint32_t e0 = __builtin_amdgcn_perm(d, d, 0x0C010C00u); // {00,b1,00,b0}
                uint32_t e1 = __builtin_amdgcn_perm(d, d, 0x0C030C02u); // {00,b3,00,b2}
                B.u[2 * j + 0] = __builtin_bit_cast(uint32_t, cvt2(e0));
                B.u[2 * j + 1] = __builtin_bit_cast(uint32_t, cvt2(e1));
            }
            const v16h bf = B.v;

            #pragma unroll
            for (int m = 0; m < 4; ++m) {
                // A fragment: row (16m + lane%16), two contiguous 16B runs (contiguous order k = 16*(lane/16) + s, same as B)
                const _Float16* ap = xh + (size_t)(m * 16 + col) * IN_F + k0 + aoff;
                union { struct { v8h lo; v8h hi; } p; v16h v; } A;
                A.p.lo = *(const v8h*)ap;
                A.p.hi = *(const v8h*)(ap + 8);

                gacc[m] = __builtin_amdgcn_wmma_f32_16x16x32_f16(
                    /*neg_a=*/false, A.v, /*neg_b=*/false, bf,
                    /*c_mod=*/(short)0, gacc[m],
                    /*reuse_a=*/false, /*reuse_b=*/false);
            }
        }

        // Exact group-wise dequant: acc += scale[o,g] * (x . wq) in fp32
        const float sc = (float)srow[g];
        #pragma unroll
        for (int m = 0; m < 4; ++m) {
            #pragma unroll
            for (int e = 0; e < 8; ++e) acc[m][e] = fmaf(sc, gacc[m][e], acc[m][e]);
        }
    }

    // Deterministic cross-wave K reduction via LDS (fixed order ((p0+p1)+p2)+p3).
    // Row stride 36 floats keeps 16B alignment for b128 while spreading banks.
    __shared__ float red[3][32][36];
    if (wave != 0) {
        #pragma unroll
        for (int m = 0; m < 4; ++m)
            *(v8f*)&red[wave - 1][lane][m * 8] = acc[m];
    }
    __syncthreads();

    if (wave == 0) {
        #pragma unroll
        for (int w = 0; w < 3; ++w) {
            #pragma unroll
            for (int m = 0; m < 4; ++m) {
                v8f p = *(const v8f*)&red[w][lane][m * 8];
                #pragma unroll
                for (int e = 0; e < 8; ++e) acc[m][e] += p[e];
            }
        }

        const float bo = (float)bias[o] + (float)resid[o];
        #pragma unroll
        for (int m = 0; m < 4; ++m) {
            const int r0 = m * 16 + hi * 8;  // C layout: M = e + 8*(lane/16)
            #pragma unroll
            for (int e = 0; e < 8; ++e)
                out[(size_t)(r0 + e) * OUT_F + o] = acc[m][e] + bo;
        }
    }
}

extern "C" void kernel_launch(void* const* d_in, const int* in_sizes, int n_in,
                              void* d_out, int out_size, void* d_ws, size_t ws_size,
                              hipStream_t stream) {
    (void)in_sizes; (void)n_in; (void)out_size; (void)ws_size;
    const float*    x     = (const float*)d_in[0];
    const uint8_t*  wq    = (const uint8_t*)d_in[1];
    const _Float16* scale = (const _Float16*)d_in[2];
    const _Float16* bias  = (const _Float16*)d_in[3];
    const _Float16* resid = (const _Float16*)d_in[4];
    float*          out   = (float*)d_out;
    _Float16*       xh    = (_Float16*)d_ws;  // 64*4096*2 = 512 KB scratch

    const int n = MB * IN_F;
    xcvt_kernel<<<(n + 255) / 256, 256, 0, stream>>>(x, xh, n);
    qlinear_wmma_kernel<<<OUT_F / 16, 128, 0, stream>>>(xh, wq, scale, bias, resid, out);
}
